// Block_56676388438228
// MI455X (gfx1250) — compile-verified, matrix operand order checked by reading, not run on hardware
//
#include <hip/hip_runtime.h>
#include <hip/hip_bf16.h>
#include <math.h>

// ---------------------------------------------------------------------------
// Types for CDNA5 WMMA (wave32)
// ---------------------------------------------------------------------------
typedef __attribute__((ext_vector_type(16))) __bf16 bf16x16;
typedef __attribute__((ext_vector_type(8)))  __bf16 bf16x8;
typedef __attribute__((ext_vector_type(8)))  float  f32x8;
typedef __attribute__((ext_vector_type(4)))  unsigned int u32x4;
typedef __attribute__((ext_vector_type(8)))  int    i32x8;
typedef __attribute__((ext_vector_type(4)))  int    i32x4;

// Async-load pointee type must match the builtin signature exactly:
// 'int __attribute__((__vector_size__(16)))' in AS1 (global) / AS3 (LDS).
typedef int v4i_b128 __attribute__((__vector_size__(16)));
typedef __attribute__((address_space(1))) v4i_b128 g_b128_t;
typedef __attribute__((address_space(3))) v4i_b128 l_b128_t;
typedef __attribute__((address_space(3))) void lds_void_t;

static_assert(sizeof(__bf16) == 2, "bf16 size");

// ---- CDNA5 data-mover feature detection (device pass only) ----------------
#if defined(__HIP_DEVICE_COMPILE__) && \
    __has_builtin(__builtin_amdgcn_global_load_async_to_lds_b128) && \
    __has_builtin(__builtin_amdgcn_s_wait_asynccnt)
#define HAVE_ASYNC_LDS 1
#else
#define HAVE_ASYNC_LDS 0
#endif

#if defined(__HIP_DEVICE_COMPILE__) && \
    __has_builtin(__builtin_amdgcn_tensor_load_to_lds) && \
    __has_builtin(__builtin_amdgcn_s_wait_tensorcnt)
#define HAVE_TDM 1
#else
#define HAVE_TDM 0
#endif

// 16B global->LDS copy: async (no VGPR round-trip, ASYNCcnt) when available.
__device__ static inline void copy16_g2l(void* dst_lds, const void* src_glob) {
#if HAVE_ASYNC_LDS
    __builtin_amdgcn_global_load_async_to_lds_b128(
        (g_b128_t*)src_glob, (l_b128_t*)dst_lds, 0, 0);
#else
    *(bf16x8*)dst_lds = *(const bf16x8*)src_glob;
#endif
}

__device__ static inline void async_fence() {
#if HAVE_ASYNC_LDS
    __builtin_amdgcn_s_wait_asynccnt(0);
#endif
}

__device__ static inline f32x8 wmma_bf16(bf16x16 a, bf16x16 b, f32x8 c) {
    // D = A(16x32) * B(32x16) + C(16x16), fp32 accum
    return __builtin_amdgcn_wmma_f32_16x16x32_bf16(
        /*neg_a=*/false, a, /*neg_b=*/false, b,
        /*c_mod=*/(short)0, c, /*reuse_a=*/false, /*reuse_b=*/false);
}

__device__ static inline f32x8 zero8() {
    f32x8 z = {0.f, 0.f, 0.f, 0.f, 0.f, 0.f, 0.f, 0.f};
    return z;
}

// Load a 16x32 bf16 fragment from LDS laid out row-major [row][k], row stride
// `stride` elements, starting at (row0, kcol). Per ISA 7.12.2 (16-bit A 16x32):
// lane&15 = row, lane>>4 selects k-half; elems j=0..7 -> k=kh*8+j,
// j=8..15 -> k=16+kh*8+(j-8). Two contiguous 16B LDS loads.
__device__ static inline bf16x16 load_frag(const __bf16* base, int row0,
                                           int stride, int kcol, int lane) {
    const __bf16* p = base + (size_t)(row0 + (lane & 15)) * stride
                           + kcol + (lane >> 4) * 8;
    bf16x16 f;
    *((bf16x8*)&f)     = *(const bf16x8*)(p);
    *(((bf16x8*)&f)+1) = *(const bf16x8*)(p + 16);
    return f;
}

// ---------------------------------------------------------------------------
// Elementwise helpers
// ---------------------------------------------------------------------------
__global__ __launch_bounds__(256) void cvt_kernel(const float* __restrict__ in,
                                                  __bf16* __restrict__ out, int n) {
    int i = blockIdx.x * 256 + threadIdx.x;
    if (i < n) out[i] = (__bf16)in[i];
}

// Broadcast memories (64x768 f32) into every (a,b) slot of x_kv rows 256..319
__global__ __launch_bounds__(256) void memfill_kernel(const float* __restrict__ mem,
                                                      __bf16* __restrict__ xkv) {
    int i   = blockIdx.x * 256 + threadIdx.x;   // 32*64*768 total
    int c   = i % 768;
    int j   = (i / 768) % 64;
    int bbs = i / (768 * 64);
    xkv[((size_t)bbs * 320 + 256 + j) * 768 + c] = (__bf16)mem[j * 768 + c];
}

// LayerNorm over 768, write bf16; optional second copy into x_kv layout
// (row -> (row/256)*320 + row%256).
__global__ __launch_bounds__(256)
void ln_kernel(const float* __restrict__ x, const float* __restrict__ g,
               const float* __restrict__ be, __bf16* __restrict__ out0,
               __bf16* __restrict__ out1) {
    __shared__ float red[256];
    int row = blockIdx.x, t = threadIdx.x;
    const float* xr = x + (size_t)row * 768;
    float v0 = xr[t], v1 = xr[t + 256], v2 = xr[t + 512];
    red[t] = v0 + v1 + v2;
    __syncthreads();
    for (int o = 128; o > 0; o >>= 1) { if (t < o) red[t] += red[t + o]; __syncthreads(); }
    float mean = red[0] * (1.0f / 768.0f);
    __syncthreads();
    float d0 = v0 - mean, d1 = v1 - mean, d2 = v2 - mean;
    red[t] = d0 * d0 + d1 * d1 + d2 * d2;
    __syncthreads();
    for (int o = 128; o > 0; o >>= 1) { if (t < o) red[t] += red[t + o]; __syncthreads(); }
    float rstd = rsqrtf(red[0] * (1.0f / 768.0f) + 1e-5f);

    __bf16* o0 = out0 + (size_t)row * 768;
    __bf16* o1 = out1 ? out1 + ((size_t)(row >> 8) * 320 + (row & 255)) * 768 : nullptr;
    float y;
    y = d0 * rstd * g[t]       + be[t];       o0[t]       = (__bf16)y; if (o1) o1[t]       = (__bf16)y;
    y = d1 * rstd * g[t + 256] + be[t + 256]; o0[t + 256] = (__bf16)y; if (o1) o1[t + 256] = (__bf16)y;
    y = d2 * rstd * g[t + 512] + be[t + 512]; o0[t + 512] = (__bf16)y; if (o1) o1[t + 512] = (__bf16)y;
}

// ---------------------------------------------------------------------------
// Generic WMMA GEMM: C(MxN) = A(MxK,row-major bf16) * B(KxN,row-major bf16)
// Block tile 128x128x32, 256 threads = 8 waves, each wave owns 2x4 16x16 tiles.
// A tile staged by the Tensor Data Mover (one TENSOR_LOAD_TO_LDS per k-step,
// TENSORcnt) when available, else async-to-LDS, else manual.
// EPI 0: outh = bf16(acc*alpha)
// EPI 1: outf = resid + acc + bias    (fp32)
// EPI 2: outh = bf16(gelu(acc+bias))  (exact GELU)
// ---------------------------------------------------------------------------
template <int EPI>
__global__ __launch_bounds__(256)
void gemm_wmma(const __bf16* __restrict__ A, const __bf16* __restrict__ B,
               int M, int N, int K,
               const float* __restrict__ bias, const float* __restrict__ resid,
               float alpha, __bf16* __restrict__ outh, float* __restrict__ outf) {
    __shared__ __align__(16) __bf16 aT[128 * 32];  // [m][k]
    __shared__ __align__(16) __bf16 bT[128 * 32];  // [n][k] (transposed)

    int tid  = threadIdx.x;
    int lane = tid & 31, wv = tid >> 5;
    int m0 = blockIdx.y * 128, n0 = blockIdx.x * 128;
    int wm = wv >> 1;   // 0..3 : mtiles {2*wm, 2*wm+1}
    int wn = wv & 1;    // 0..1 : ntiles {4*wn .. 4*wn+3}

    f32x8 acc[2][4];
#pragma unroll
    for (int i = 0; i < 2; ++i)
#pragma unroll
        for (int j = 0; j < 4; ++j) acc[i][j] = zero8();

#if HAVE_TDM
    // TDM descriptor group1 (ISA 8.4): 2-byte elements, 2-D tensor M x K,
    // tile 128 rows x 32 cols, row stride K. Invariant across k-steps.
    i32x8 g1;
    g1[0] = 0x00010000;                                    // data_size = 1 (2B)
    g1[1] = (int)((unsigned)(K & 0xFFFF) << 16);           // tensor_dim0[15:0]
    g1[2] = (int)((((unsigned)K >> 16) & 0xFFFFu) |        // tensor_dim0[31:16]
                  (((unsigned)M & 0xFFFFu) << 16));        // tensor_dim1[15:0]
    g1[3] = (int)(((((unsigned)M) >> 16) & 0xFFFFu) |      // tensor_dim1[31:16]
                  (32u << 16));                            // tile_dim0 = 32
    g1[4] = 128;                                           // tile_dim1 = 128
    g1[5] = K;                                             // dim0 stride [31:0]
    g1[6] = 0;                                             // stride [47:32], dim1 stride lo
    g1[7] = 0;
    i32x4 gz = {0, 0, 0, 0};
    unsigned aT_lds = (unsigned)(unsigned long long)(lds_void_t*)aT;
#endif

    for (int k0 = 0; k0 < K; k0 += 32) {
        __syncthreads();
#if HAVE_TDM
        if (wv == 0) {
            unsigned long long ga =
                (unsigned long long)(const void*)(A + (size_t)m0 * K + k0);
            u32x4 g0 = {1u,                                // count=1, user mode
                        aT_lds,                            // lds_addr
                        (unsigned)ga,                      // global_addr[31:0]
                        (unsigned)((ga >> 32) & 0x01FFFFFFull) | 0x80000000u}; // [56:32] | type=2
#if __clang_major__ >= 23
            i32x8 gz8 = {0, 0, 0, 0, 0, 0, 0, 0};
            __builtin_amdgcn_tensor_load_to_lds(g0, g1, gz, gz, gz8, 0);
#else
            __builtin_amdgcn_tensor_load_to_lds(g0, g1, gz, gz, 0);
#endif
        }
#else
        // Stage A tile 128x32 (two 16B copies per thread)
#pragma unroll
        for (int s = 0; s < 2; ++s) {
            int lin = tid + s * 256;
            int m   = lin >> 2;
            int kk  = (lin & 3) * 8;
            copy16_g2l(&aT[m * 32 + kk], &A[(size_t)(m0 + m) * K + k0 + kk]);
        }
#endif
        // Stage B tile 32x128, transposed into [n][k] (needs VGPR round-trip)
#pragma unroll
        for (int s = 0; s < 2; ++s) {
            int lin = tid + s * 256;
            int kk  = lin >> 4;
            int nn  = (lin & 15) * 8;
            bf16x8 v = *(const bf16x8*)&B[(size_t)(k0 + kk) * N + n0 + nn];
#pragma unroll
            for (int i = 0; i < 8; ++i) bT[(nn + i) * 32 + kk] = v[i];
        }
#if HAVE_TDM
        if (wv == 0) __builtin_amdgcn_s_wait_tensorcnt(0);
#else
        async_fence();
#endif
        __syncthreads();

        bf16x16 af[2];
#pragma unroll
        for (int i = 0; i < 2; ++i)
            af[i] = load_frag(aT, (wm * 2 + i) * 16, 32, 0, lane);
#pragma unroll
        for (int j = 0; j < 4; ++j) {
            bf16x16 bf = load_frag(bT, (wn * 4 + j) * 16, 32, 0, lane);
#pragma unroll
            for (int i = 0; i < 2; ++i) acc[i][j] = wmma_bf16(af[i], bf, acc[i][j]);
        }
    }

    // Epilogue: C layout — VGPR r, lane l: M = (l>>4)*8 + r, N = l&15
    int rlo  = (lane >> 4) * 8;
    int ncol = lane & 15;
#pragma unroll
    for (int i = 0; i < 2; ++i)
#pragma unroll
        for (int j = 0; j < 4; ++j) {
            int mr = m0 + (wm * 2 + i) * 16 + rlo;
            int nc = n0 + (wn * 4 + j) * 16 + ncol;
#pragma unroll
            for (int r = 0; r < 8; ++r) {
                size_t idx = (size_t)(mr + r) * N + nc;
                float v = acc[i][j][r];
                if (EPI == 0) {
                    outh[idx] = (__bf16)(v * alpha);
                } else if (EPI == 1) {
                    outf[idx] = resid[idx] + v + bias[nc];
                } else {
                    float u = v + bias[nc];
                    outh[idx] = (__bf16)(0.5f * u * (1.0f + erff(u * 0.70710678118f)));
                }
            }
        }
}

// ---------------------------------------------------------------------------
// Attention: one block per (a,b,h, 64-query slab). LDS-resident 64x320 scores.
// Q/K staged with async-to-LDS; V transposed manually. S = (Q*scale) K^T ->
// mask -> softmax -> P V -> O written into the reference's swapaxes(1,2)
// layout: contiguous (a, h, b, n, d).
// ---------------------------------------------------------------------------
__global__ __launch_bounds__(256)
void attn_kernel(const __bf16* __restrict__ q, const __bf16* __restrict__ kv,
                 const unsigned char* __restrict__ mask, __bf16* __restrict__ o) {
    extern __shared__ __align__(16) char smem[];
    __bf16* q_s  = (__bf16*)smem;             // 64 x 64
    __bf16* k_s  = q_s + 64 * 64;             // 320 x 64 (row-major = B^T-major)
    __bf16* vt_s = k_s + 320 * 64;            // 64 x 320 (V transposed: [d][key])
    float*  s_s  = (float*)(vt_s + 64 * 320); // 64 x 320 scores (fp32)
    __bf16* p_s  = (__bf16*)(s_s + 64 * 320); // 64 x 320 probabilities (bf16)

    int tid = threadIdx.x, lane = tid & 31, w = tid >> 5;
    int idx = blockIdx.x;            // (a*8+b)*12 + h
    int h   = idx % 12;
    int bb  = idx / 12;              // a*8 + b
    int a   = bb >> 3, b = bb & 7;
    int qb  = blockIdx.y;            // 0..3
    int tok0 = bb * 256 + qb * 64;

    // Stage Q (64x64) — async to LDS
#pragma unroll
    for (int s = 0; s < 2; ++s) {
        int lin = tid + s * 256;
        int m = lin >> 3, d = (lin & 7) * 8;
        copy16_g2l(&q_s[m * 64 + d], &q[(size_t)(tok0 + m) * 768 + h * 64 + d]);
    }
    // Stage K (320x64, async) and V^T (64x320, manual transpose)
#pragma unroll
    for (int s = 0; s < 10; ++s) {
        int lin = tid + s * 256;
        int j = lin >> 3, d = (lin & 7) * 8;
        const __bf16* kvrow = &kv[(size_t)(bb * 320 + j) * 1536];
        copy16_g2l(&k_s[j * 64 + d], &kvrow[h * 64 + d]);
        bf16x8 vv = *(const bf16x8*)&kvrow[768 + h * 64 + d];
#pragma unroll
        for (int i = 0; i < 8; ++i) vt_s[(d + i) * 320 + j] = vv[i];
    }
    async_fence();
    __syncthreads();

    // ---- S = Q K^T (scale pre-folded into q), mask, store fp32 to LDS ----
    {
        int mt = w >> 1;
        bf16x16 a0 = load_frag(q_s, mt * 16, 64, 0, lane);
        bf16x16 a1 = load_frag(q_s, mt * 16, 64, 32, lane);
        size_t mbase = ((size_t)(bb * 12 + h) * 256 + qb * 64) * 320;
        const unsigned char* mrow = mask + mbase;
        for (int jt = (w & 1); jt < 20; jt += 2) {
            f32x8 c = zero8();
            bf16x16 b0 = load_frag(k_s, jt * 16, 64, 0, lane);
            bf16x16 b1 = load_frag(k_s, jt * 16, 64, 32, lane);
            c = wmma_bf16(a0, b0, c);
            c = wmma_bf16(a1, b1, c);
            int col   = jt * 16 + (lane & 15);
            int rbase = mt * 16 + (lane >> 4) * 8;
#pragma unroll
            for (int r = 0; r < 8; ++r) {
                int row = rbase + r;
                float sc = mrow[(size_t)row * 320 + col] ? c[r] : -3.0e38f;
                s_s[row * 320 + col] = sc;
            }
        }
    }
    __syncthreads();

    // ---- softmax over 320 keys: 4 lanes per row (quad shuffles, wave32) ----
    {
        int row = tid >> 2, part = tid & 3;
        float* sr = &s_s[row * 320 + part * 80];
        float mx = -3.4e38f;
        for (int i = 0; i < 80; ++i) mx = fmaxf(mx, sr[i]);
        mx = fmaxf(mx, __shfl_xor(mx, 1, 4));
        mx = fmaxf(mx, __shfl_xor(mx, 2, 4));
        float sum = 0.f;
        for (int i = 0; i < 80; ++i) { float e = __expf(sr[i] - mx); sr[i] = e; sum += e; }
        sum += __shfl_xor(sum, 1, 4);
        sum += __shfl_xor(sum, 2, 4);
        float inv = 1.0f / sum;
        __bf16* pr = &p_s[row * 320 + part * 80];
        for (int i = 0; i < 80; ++i) pr[i] = (__bf16)(sr[i] * inv);
    }
    __syncthreads();

    // ---- O = P(64x320) V(320x64) ----
    {
        int mt = w >> 1;
        int nt0 = (w & 1) * 2;
        f32x8 c[2] = {zero8(), zero8()};
        for (int ks = 0; ks < 10; ++ks) {
            bf16x16 af = load_frag(p_s, mt * 16, 320, ks * 32, lane);
#pragma unroll
            for (int j = 0; j < 2; ++j) {
                bf16x16 bf = load_frag(vt_s, (nt0 + j) * 16, 320, ks * 32, lane);
                c[j] = wmma_bf16(af, bf, c[j]);
            }
        }
        int rbase = mt * 16 + (lane >> 4) * 8;
#pragma unroll
        for (int j = 0; j < 2; ++j) {
            int d = (nt0 + j) * 16 + (lane & 15);
#pragma unroll
            for (int r = 0; r < 8; ++r) {
                int n = qb * 64 + rbase + r;
                // faithful swapaxes(1,2): contiguous (a, h, b, n, d)
                size_t off = ((((size_t)a * 12 + h) * 8 + b) * 256 + n) * 64 + d;
                o[off] = (__bf16)c[j][r];
            }
        }
    }
}

// ---------------------------------------------------------------------------
// Host side
// ---------------------------------------------------------------------------
extern "C" void kernel_launch(void* const* d_in, const int* in_sizes, int n_in,
                              void* d_out, int out_size, void* d_ws, size_t ws_size,
                              hipStream_t stream) {
    const float* x     = (const float*)d_in[0];
    const unsigned char* mask = (const unsigned char*)d_in[1];
    const float* mem   = (const float*)d_in[2];
    const float* wq    = (const float*)d_in[3];
    const float* wkv   = (const float*)d_in[4];
    const float* wproj = (const float*)d_in[5];
    const float* bproj = (const float*)d_in[6];
    const float* w1    = (const float*)d_in[7];
    const float* b1    = (const float*)d_in[8];
    const float* w2    = (const float*)d_in[9];
    const float* b2    = (const float*)d_in[10];
    const float* g1    = (const float*)d_in[11];
    const float* be1   = (const float*)d_in[12];
    const float* g2    = (const float*)d_in[13];
    const float* be2   = (const float*)d_in[14];

    char* ws = (char*)d_ws;
    size_t off = 0;
    auto alloc = [&](size_t bytes) {
        void* p = ws + off;
        off += (bytes + 255) & ~(size_t)255;
        return p;
    };
    __bf16* wq_h    = (__bf16*)alloc((size_t)768 * 768 * 2);
    __bf16* wkv_h   = (__bf16*)alloc((size_t)768 * 1536 * 2);
    __bf16* wproj_h = (__bf16*)alloc((size_t)768 * 768 * 2);
    __bf16* w1_h    = (__bf16*)alloc((size_t)768 * 3072 * 2);
    __bf16* w2_h    = (__bf16*)alloc((size_t)3072 * 768 * 2);
    __bf16* xn_h    = (__bf16*)alloc((size_t)8192 * 768 * 2);
    __bf16* xkv_h   = (__bf16*)alloc((size_t)10240 * 768 * 2);
    __bf16* q_h     = (__bf16*)alloc((size_t)8192 * 768 * 2);
    __bf16* kv_h    = (__bf16*)alloc((size_t)10240 * 1536 * 2);
    __bf16* o_h     = (__bf16*)alloc((size_t)8192 * 768 * 2);
    float*  x2      = (float*)alloc((size_t)8192 * 768 * 4);
    __bf16* hn_h    = (__bf16*)alloc((size_t)8192 * 768 * 2);
    __bf16* h1_h    = (__bf16*)alloc((size_t)8192 * 3072 * 2);

    auto cvt = [&](const float* src, __bf16* dst, int n) {
        cvt_kernel<<<(n + 255) / 256, 256, 0, stream>>>(src, dst, n);
    };
    cvt(wq, wq_h, 768 * 768);
    cvt(wkv, wkv_h, 768 * 1536);
    cvt(wproj, wproj_h, 768 * 768);
    cvt(w1, w1_h, 768 * 3072);
    cvt(w2, w2_h, 3072 * 768);

    // LN1 -> xn (bf16) and x_kv rows 0..255 of each slab
    ln_kernel<<<8192, 256, 0, stream>>>(x, g1, be1, xn_h, xkv_h);
    // memories -> x_kv rows 256..319 of each slab
    memfill_kernel<<<32 * 64 * 768 / 256, 256, 0, stream>>>(mem, xkv_h);

    // Q = xn @ wq  (fold softmax scale 1/8 here)
    gemm_wmma<0><<<dim3(768 / 128, 8192 / 128), 256, 0, stream>>>(
        xn_h, wq_h, 8192, 768, 768, nullptr, nullptr, 0.125f, q_h, nullptr);
    // KV = x_kv @ wkv
    gemm_wmma<0><<<dim3(1536 / 128, 10240 / 128), 256, 0, stream>>>(
        xkv_h, wkv_h, 10240, 1536, 768, nullptr, nullptr, 1.0f, kv_h, nullptr);

    // Attention: 384 (a,b,h) blocks x 4 query slabs; 208KB dynamic LDS
    size_t attn_lds = (size_t)(64 * 64 + 320 * 64 + 64 * 320) * 2
                    + (size_t)64 * 320 * 4 + (size_t)64 * 320 * 2;
    attn_kernel<<<dim3(384, 4), 256, attn_lds, stream>>>(q_h, kv_h, mask, o_h);

    // x2 = x + (o @ wproj + bproj)
    gemm_wmma<1><<<dim3(768 / 128, 8192 / 128), 256, 0, stream>>>(
        o_h, wproj_h, 8192, 768, 768, bproj, x, 1.0f, nullptr, x2);

    // LN2
    ln_kernel<<<8192, 256, 0, stream>>>(x2, g2, be2, hn_h, nullptr);

    // h1 = gelu(hn @ w1 + b1)
    gemm_wmma<2><<<dim3(3072 / 128, 8192 / 128), 256, 0, stream>>>(
        hn_h, w1_h, 8192, 3072, 768, b1, nullptr, 1.0f, h1_h, nullptr);

    // out = x2 + (h1 @ w2 + b2)
    gemm_wmma<1><<<dim3(768 / 128, 8192 / 128), 256, 0, stream>>>(
        h1_h, w2_h, 8192, 768, 3072, b2, x2, 1.0f, nullptr, (float*)d_out);

    (void)in_sizes; (void)n_in; (void)out_size; (void)ws_size;
}
